// CIN_6588479832456
// MI455X (gfx1250) — matrix operand order repaired and checked by reading, not compiled, not run on hardware
//
#include <hip/hip_runtime.h>

// ---------------------------------------------------------------------------
// CIN (xDeepFM) on gfx1250: per-batch GEMM  Out_b[128x16] = Wpack[128xK] * Z_b[Kx16]
// with Z generated on the fly from x0/xk, computed with v_wmma_f32_16x16x32_bf16.
// Each wave processes NB=2 batch elements so every A-operand tile feeds 2 WMMAs.
// ---------------------------------------------------------------------------

typedef __attribute__((ext_vector_type(16))) __bf16 v16bf;
typedef __attribute__((ext_vector_type(8)))  float  v8f;

#define B_N 8192
#define M_F 39
#define D_E 16

// K tiling: k = m*HPAD + h, HPAD multiple of 32 so each 32-wide tile has fixed m.
// layer1: HPAD=64  (h<39 valid), T1 = 39*64/32  = 78 tiles
// layer2: HPAD=128 (h<128),      T2 = 39*128/32 = 156 tiles

// workspace layout (bytes)
#define SZ_A1   (8*78*32*16*2)          // 638976   W1 packed (A-operand swizzle)
#define SZ_A2   (8*156*32*16*2)         // 1277952  W2 packed
#define SZ_X0T  ((size_t)B_N*16*40*2)   // x0 transposed [b][d][m(40, pad0)] bf16
#define SZ_X1T  ((size_t)B_N*16*128*2)  // x1 transposed [b][d][s=128] bf16
#define OFF_A1  ((size_t)0)
#define OFF_A2  (OFF_A1 + SZ_A1)
#define OFF_X0T (OFF_A2 + SZ_A2)
#define OFF_X1T (OFF_X0T + SZ_X0T)

// ---- repack x0 [B,M,D] f32 -> x0T [B][d][m] bf16 (m padded to 40 with zeros)
__global__ void cin_repack_x0(const float* __restrict__ x0, __bf16* __restrict__ x0T) {
    int idx = blockIdx.x * 256 + threadIdx.x;           // B*640 total
    int b = idx / 640, rem = idx % 640;
    int d = rem / 40,  m = rem % 40;
    float v = (m < M_F) ? x0[(size_t)b * (M_F * D_E) + m * D_E + d] : 0.0f;
    x0T[idx] = (__bf16)v;
}

// ---- repack W [S, H, M] f32 -> A-operand swizzled bf16: [r][t][lane][e]
// A tile, this code's contiguous k order (B matches): lane<16 holds M=lane, K = e;
// lane>=16 holds M=lane-16, K = e + 16.
template<int TT, int HSHIFT, int HVALID, int SSTRIDE>
__global__ void cin_repack_w(const float* __restrict__ W, __bf16* __restrict__ Apack) {
    int idx  = blockIdx.x * 256 + threadIdx.x;          // 8*TT*512 total
    int e    = idx & 15;
    int lane = (idx >> 4) & 31;
    int rt   = idx >> 9;
    int t    = rt % TT, r = rt / TT;
    int koff = e + ((lane & 16) ? 16 : 0);
    int k    = t * 32 + koff;
    int m    = k >> HSHIFT;
    int h    = k & ((1 << HSHIFT) - 1);
    int s    = r * 16 + (lane & 15);
    float v  = (h < HVALID) ? W[(size_t)s * SSTRIDE + h * M_F + m] : 0.0f;
    Apack[idx] = (__bf16)v;
}

// ---- fused CIN layer: one wave per 2 batch elements, 8 WMMA row-tiles (S=128)
template<int HPAD, int TT, int OUT_OFF, bool WRITE_X1, bool PAD_XK>
__global__ __launch_bounds__(128)
void cin_layer(const __bf16* __restrict__ Apack,
               const __bf16* __restrict__ xkT,   // PAD_XK: x0T [b][d][40]; else x1T [b][d][HPAD]
               const __bf16* __restrict__ x0T,   // [b][d][40]
               const float*  __restrict__ bias,  // [128]
               float*        __restrict__ out,   // [B][256]
               __bf16*       __restrict__ x1T)   // [B][d][128] (layer1 only)
{
    constexpr int TPM_SHIFT     = (HPAD == 64) ? 1 : 2;  // tiles per m = HPAD/32
    constexpr int LDS_PER_BATCH = 16 * HPAD + 16 * 40;
    __shared__ __bf16 lds[4 * 2 * LDS_PER_BATCH];

    const int lane = threadIdx.x & 31;
    const int wave = threadIdx.x >> 5;
    const int b0   = blockIdx.x * 8 + wave * 2;          // this wave: batches b0, b0+1

    __bf16* xk[2];
    __bf16* x0l[2];
    #pragma unroll
    for (int n = 0; n < 2; ++n) {
        xk[n]  = &lds[(wave * 2 + n) * LDS_PER_BATCH];   // [d][HPAD], zero-padded h
        x0l[n] = xk[n] + 16 * HPAD;                      // [d][40]
        const int b = b0 + n;
        if constexpr (PAD_XK) {                          // layer1: xk = x0, pad h to 64
            for (int i = lane; i < 16 * HPAD; i += 32) {
                int d = i / HPAD, h = i % HPAD;
                xk[n][i] = (h < 40) ? xkT[(size_t)b * 640 + d * 40 + h] : (__bf16)0.0f;
            }
        } else {                                         // layer2: xk = x1 ([d][128])
            for (int i = lane; i < 16 * HPAD; i += 32)
                xk[n][i] = xkT[(size_t)b * (16 * HPAD) + i];
        }
        for (int i = lane; i < 640; i += 32)
            x0l[n][i] = x0T[(size_t)b * 640 + i];
    }

    const int d   = lane & 15;                           // B/C/D operand: N = lane&15
    const int sel = lane >> 4;

    v8f acc[2][8];
    #pragma unroll
    for (int n = 0; n < 2; ++n)
        #pragma unroll
        for (int r = 0; r < 8; ++r) { v8f z = {}; acc[n][r] = z; }

    for (int t = 0; t < TT; ++t) {
        const int m  = t >> TPM_SHIFT;
        const int h0 = (t & ((1 << TPM_SHIFT) - 1)) * 32;

        // B-operands: lane holds K = h0 + sel*16 + e -> 16 contiguous xk[h] values
        v16bf bv[2];
        #pragma unroll
        for (int n = 0; n < 2; ++n) {
            v16bf xv = *reinterpret_cast<const v16bf*>(&xk[n][d * HPAD + h0 + sel * 16]);
            float x0f = (float)x0l[n][d * 40 + m];
            #pragma unroll
            for (int e = 0; e < 16; ++e)
                bv[n][e] = (__bf16)((float)xv[e] * x0f);
        }

        const v16bf* ap = reinterpret_cast<const v16bf*>(Apack) + (size_t)t * 32 + lane;
        #pragma unroll
        for (int r = 0; r < 8; ++r) {
            v16bf av = ap[(size_t)r * TT * 32];          // one A tile feeds 2 WMMAs
            acc[0][r] = __builtin_amdgcn_wmma_f32_16x16x32_bf16(
                            false, av, false, bv[0], (short)0, acc[0][r], false, false);
            acc[1][r] = __builtin_amdgcn_wmma_f32_16x16x32_bf16(
                            false, av, false, bv[1], (short)0, acc[1][r], false, false);
        }
    }

    // epilogue: bias + ReLU, write x1 (bf16, d-major), reduce over d for pooled sum
    #pragma unroll
    for (int n = 0; n < 2; ++n) {
        const int b = b0 + n;
        #pragma unroll
        for (int r = 0; r < 8; ++r) {
            #pragma unroll
            for (int v = 0; v < 8; ++v) {
                const int s = r * 16 + sel * 8 + v;      // C/D layout: VGPR v -> row v+sel*8
                float val = acc[n][r][v] + bias[s];
                val = fmaxf(val, 0.0f);
                if constexpr (WRITE_X1)
                    x1T[(size_t)b * 2048 + d * 128 + s] = (__bf16)val;
                float sum = val;
                #pragma unroll
                for (int msk = 1; msk < 16; msk <<= 1)
                    sum += __shfl_xor(sum, msk, 32);
                if (d == 0)
                    out[(size_t)b * 256 + OUT_OFF + s] = sum;
            }
        }
    }
}

extern "C" void kernel_launch(void* const* d_in, const int* in_sizes, int n_in,
                              void* d_out, int out_size, void* d_ws, size_t ws_size,
                              hipStream_t stream) {
    const float* x0 = (const float*)d_in[0];
    const float* W1 = (const float*)d_in[1];
    const float* b1 = (const float*)d_in[2];
    const float* W2 = (const float*)d_in[3];
    const float* b2 = (const float*)d_in[4];
    float* out = (float*)d_out;

    char* ws = (char*)d_ws;
    __bf16* A1  = (__bf16*)(ws + OFF_A1);
    __bf16* A2  = (__bf16*)(ws + OFF_A2);
    __bf16* x0T = (__bf16*)(ws + OFF_X0T);
    __bf16* x1T = (__bf16*)(ws + OFF_X1T);

    cin_repack_x0<<<(B_N * 640) / 256, 256, 0, stream>>>(x0, x0T);
    cin_repack_w<78, 6, 39, 39 * 39>  <<<(8 * 78 * 512) / 256, 256, 0, stream>>>(W1, A1);
    cin_repack_w<156, 7, 128, 128 * 39><<<(8 * 156 * 512) / 256, 256, 0, stream>>>(W2, A2);

    // layer1: xk = x0 (padded to HPAD=64), writes x1T + p1 (cols 0..127)
    cin_layer<64, 78, 0, true, true>
        <<<B_N / 8, 128, 0, stream>>>(A1, x0T, x0T, b1, out, x1T);
    // layer2: xk = x1 (HPAD=128), writes p2 (cols 128..255)
    cin_layer<128, 156, 128, false, false>
        <<<B_N / 8, 128, 0, stream>>>(A2, x1T, x0T, b2, out, nullptr);
}
